// EfficientHashedLinear_72043781423546
// MI455X (gfx1250) — matrix operand order repaired and checked by reading, not compiled, not run on hardware
//
#include <hip/hip_runtime.h>
#include <hip/hip_bf16.h>
#include <cstdint>

#define B_DIM 64
#define D_DIM 2048
#define UNITS 1024
#define NW    64

typedef __attribute__((ext_vector_type(16))) _Float16 v16h;
typedef __attribute__((ext_vector_type(8)))  _Float16 v8h;
typedef __attribute__((ext_vector_type(8)))  float    v8f;

// ---- 1) zero-init W_eff (f16, D_DIM x UNITS), written as u32 pairs ----
__global__ void ehl_zero_w(uint32_t* __restrict__ w32) {
    int t = blockIdx.x * blockDim.x + threadIdx.x;   // exactly D*U/2 threads
    w32[t] = 0u;
}

// ---- 2) x (f32) -> x (f16), row-major [B][D] so K is contiguous ----
__global__ void ehl_cvt_x(const float* __restrict__ x, _Float16* __restrict__ xh) {
    int t = blockIdx.x * blockDim.x + threadIdx.x;   // exactly B*D threads
    xh[t] = (_Float16)x[t];
}

// ---- 3) scatter: W_eff[idx-1][u] = w[k] for every (u,k,l) with idx>0 ----
__global__ void ehl_scatter_w(const int* __restrict__ idx,
                              const float* __restrict__ w,
                              _Float16* __restrict__ W, int L) {
    const int u   = blockIdx.x;          // one block per unit
    const int NWL = NW * L;
    const int* iu = idx + (size_t)u * NWL;
    for (int t = threadIdx.x; t < NWL; t += blockDim.x) {
        int id = iu[t];                  // coalesced index read
        if (id > 0) {
            int k = t / L;
            W[(size_t)(id - 1) * UNITS + u] = (_Float16)w[k];
        }
    }
}

// ---- 4) GEMM: out[64x1024] = Xh[64x2048] * W[2048x1024] + bias ----
// one wave (32 threads) per 16x16 output tile; K-loop of v_wmma_f32_16x16x32_f16
__global__ void __launch_bounds__(32)
ehl_wmma_gemm(const _Float16* __restrict__ A,   // [B_DIM][D_DIM] f16
              const _Float16* __restrict__ Wm,  // [D_DIM][UNITS] f16
              const float*    __restrict__ bias,
              float*          __restrict__ out) {
    const int lane = threadIdx.x;        // 0..31
    const int g    = lane >> 4;          // lane group 0/1
    const int mr   = lane & 15;
    const int m0   = blockIdx.x * 16;    // batch tile
    const int n0   = blockIdx.y * 16;    // unit tile

    const _Float16* arow = A  + (size_t)(m0 + mr) * D_DIM;
    const _Float16* bcol = Wm + (size_t)n0;

    v8f acc = {};
    #pragma unroll 4
    for (int k0 = 0; k0 < D_DIM; k0 += 32) {
        // A 16x32 f16 tile, ISA layout: halves 0..7 <- K in [g*8, g*8+8),
        //                               halves 8..15 <- K in [16+g*8, ...+8)
        union { v16h v; v8h h[2]; } a;
        a.h[0] = *(const v8h*)(arow + k0 + g * 8);
        a.h[1] = *(const v8h*)(arow + k0 + 16 + g * 8);
        // B 32x16 f16 tile: lane holds column N=mr, K in same order as A
        v16h bm;
        #pragma unroll
        for (int s = 0; s < 16; ++s) {
            int kk = (s < 8) ? (g * 8 + s) : (16 + g * 8 + (s - 8));
            bm[s] = bcol[(size_t)(k0 + kk) * UNITS + mr];
        }
        acc = __builtin_amdgcn_wmma_f32_16x16x32_f16(
                  /*neg_a=*/false, a.v, /*neg_b=*/false, bm,
                  /*c_mod=*/(short)0, acc, /*reuse_a=*/false, /*reuse_b=*/false);
    }

    // D layout: VGPR r -> M = r + g*8, N = mr
    const int   ncol = n0 + mr;
    const float bb   = bias[ncol];
    float* obase = out + (size_t)(m0 + g * 8) * UNITS + ncol;
    #pragma unroll
    for (int r = 0; r < 8; ++r)
        obase[(size_t)r * UNITS] = acc[r] + bb;
}

extern "C" void kernel_launch(void* const* d_in, const int* in_sizes, int n_in,
                              void* d_out, int out_size, void* d_ws, size_t ws_size,
                              hipStream_t stream) {
    const float* x   = (const float*)d_in[0];   // [64][2048]
    const float* w   = (const float*)d_in[1];   // [64]
    const float* b   = (const float*)d_in[2];   // [1024]
    const int*   idx = (const int*)d_in[3];     // [1024][64][L]
    const int L = in_sizes[3] / (UNITS * NW);

    _Float16* Wm = (_Float16*)d_ws;                                   // 4 MB
    _Float16* Xh = (_Float16*)((char*)d_ws +
                               (size_t)D_DIM * UNITS * sizeof(_Float16)); // 256 KB

    ehl_zero_w   <<<(D_DIM * UNITS / 2) / 256, 256, 0, stream>>>((uint32_t*)Wm);
    ehl_cvt_x    <<<(B_DIM * D_DIM) / 256,     256, 0, stream>>>(x, Xh);
    ehl_scatter_w<<<UNITS,                     256, 0, stream>>>(idx, w, Wm, L);
    ehl_wmma_gemm<<<dim3(B_DIM / 16, UNITS / 16), 32, 0, stream>>>(Xh, Wm, b, (float*)d_out);
}
